// GCNNet_20658792694055
// MI455X (gfx1250) — matrix operand order repaired and checked by reading, not compiled, not run on hardware
//
#include <hip/hip_runtime.h>
#include <hip/hip_bf16.h>

#define N_FIN 10
#define HIDF 128
#define NGRAPH 256
#define EPB 64   // edges staged per aggregation block

typedef __attribute__((ext_vector_type(16))) __bf16 v16bf;
typedef __attribute__((ext_vector_type(8)))  float  v8f;
typedef unsigned int u32x4 __attribute__((ext_vector_type(4)));
typedef int          i32x4 __attribute__((ext_vector_type(4)));
typedef int          i32x8 __attribute__((ext_vector_type(8)));

__device__ __forceinline__ unsigned short f2bfbits(float f) {
    union { float f; unsigned u; } in; in.f = f;
    unsigned r = in.u + 0x7FFFu + ((in.u >> 16) & 1u);   // round-to-nearest-even
    return (unsigned short)(r >> 16);
}
__device__ __forceinline__ __bf16 f2bf(float f) {
    union { unsigned short s; __bf16 b; } out;
    out.s = f2bfbits(f);
    return out.b;
}

// ---- Tensor Data Mover: 1-D bf16 DMA global -> LDS (D# per CDNA5 ISA) --------
__device__ __forceinline__ void tdm_load_bf16(unsigned lds_off, const __bf16* gsrc,
                                              unsigned tile_elems, unsigned tensor_elems) {
    unsigned long long ga = (unsigned long long)(const void*)gsrc;
    u32x4 g0;
    g0[0] = 1u;                                            // count=1, user mode
    g0[1] = lds_off;                                       // lds_addr (bytes)
    g0[2] = (unsigned)ga;                                  // global_addr[31:0]
    g0[3] = (unsigned)((ga >> 32) & 0x01FFFFFFu) | (2u << 30); // addr[56:32] | type=2
    i32x8 g1;
    g1[0] = (int)(1u << 16);                               // data_size=1 -> 2 bytes
    g1[1] = (int)((tensor_elems & 0xFFFFu) << 16);         // tensor_dim0[15:0]
    g1[2] = (int)(((tensor_elems >> 16) & 0xFFFFu) | (1u << 16)); // dim0[31:16], dim1=1
    g1[3] = (int)((tile_elems & 0xFFFFu) << 16);           // tile_dim0 (1-D tile)
    g1[4] = 0;                                             // tile_dim1=0, tile_dim2=0
    g1[5] = (int)tensor_elems;                             // dim0_stride lo32
    g1[6] = 0;
    g1[7] = 0;
    i32x4 z4 = {};
    i32x8 z8 = {};
    __builtin_amdgcn_tensor_load_to_lds(g0, g1, z4, z4, z8, 0);
}

// ---------------- degree / norm ---------------------------------------------------
__global__ void degree_kernel(const int* __restrict__ col, const float* __restrict__ w,
                              float* __restrict__ deg, long n_edges) {
    long e = (long)blockIdx.x * blockDim.x + threadIdx.x;
    if (e < n_edges) atomicAdd(&deg[col[e]], w[e]);
}
__global__ void dinv_kernel(const float* __restrict__ deg, float* __restrict__ dinv, int n) {
    int i = blockIdx.x * blockDim.x + threadIdx.x;
    if (i < n) dinv[i] = rsqrtf(deg[i] + 1.0f);   // +1 = self-loop; always > 0
}

// ---- weights -> bf16, swizzled into WMMA B-fragment layout (kb=(j<8?8h+j:16+8h+j-8)) ----
__global__ void convw_kernel(const float* __restrict__ src, __bf16* __restrict__ dst,
                             int ksrc, int kdst) {
    int i = blockIdx.x * blockDim.x + threadIdx.x;
    if (i >= kdst * HIDF) return;
    int k = i >> 7, c = i & 127;
    float v = (k < ksrc) ? src[k * HIDF + c] : 0.0f;
    int kt = k >> 5, kk = k & 31;
    int h = (kk >> 3) & 1, j = (kk & 7) + 8 * (kk >> 4);
    dst[((kt * HIDF + c) * 2 + h) * 16 + j] = f2bf(v);
}

// ---- x -> bf16 (K 10->32 pad), swizzled A-frag layout (ka=(j<8?8h+j:16+8h+j-8)) --
__global__ void convx_kernel(const float* __restrict__ x, __bf16* __restrict__ dst,
                             int n_nodes) {
    long i = (long)blockIdx.x * blockDim.x + threadIdx.x;     // node*32 + k
    if (i >= (long)n_nodes * 32) return;
    int k = (int)(i & 31);
    long node = i >> 5;
    float v = (k < N_FIN) ? x[node * N_FIN + k] : 0.0f;
    long tile = node >> 4; int m = (int)(node & 15);
    int h = (k >> 3) & 1, j = (k & 7) + 8 * (k >> 4);
    dst[tile * 512 + (m * 2 + h) * 16 + j] = f2bf(v);
}

// ---------------- GEMM1: [N,32pad] @ [32pad,128], TDM-staged, 1 WMMA / tile -------
__global__ void gemm1_kernel(const __bf16* __restrict__ Xp, const __bf16* __restrict__ Wp,
                             float* __restrict__ Hout, int n_nodes) {
    extern __shared__ char smem[];
    __bf16* Wl = (__bf16*)smem;            // 8192 B  (32x128 swizzled)
    __bf16* Al = (__bf16*)(smem + 8192);   // 1024 B  (16x32 swizzled)
    float*  Dl = (float*)(smem + 9216);    // 8192 B  (16x128 f32)
    const int tid = threadIdx.x;           // 128 threads = 4 waves
    const long node0 = (long)blockIdx.x * 16;
    if (tid < 32) {
        tdm_load_bf16(0u, Wp, 4096u, 4096u);
        tdm_load_bf16(8192u, Xp + (long)blockIdx.x * 512, 512u, 512u);
        __builtin_amdgcn_s_wait_tensorcnt((short)0);
    }
    __syncthreads();
    const int wave = tid >> 5, lane = tid & 31;
    const int half = lane >> 4, mn = lane & 15;
    v16bf a = *(const v16bf*)(Al + (mn * 2 + half) * 16);
#pragma unroll
    for (int ct = 0; ct < 2; ++ct) {
        const int c0 = (wave * 2 + ct) * 16;
        v16bf b = *(const v16bf*)(Wl + ((c0 + mn) * 2 + half) * 16);
        v8f c = {};
        c = __builtin_amdgcn_wmma_f32_16x16x32_bf16(false, a, false, b, (short)0, c, false, false);
#pragma unroll
        for (int r = 0; r < 8; ++r) Dl[(r + 8 * half) * HIDF + c0 + mn] = c[r];
    }
    __syncthreads();
#pragma unroll
    for (int it = 0; it < 4; ++it) {       // coalesced float4 stores
        int fi = it * 128 + tid;
        int m = fi >> 5, fc = (fi & 31) * 4;
        long node = node0 + m;
        if (node < n_nodes)
            *(float4*)(Hout + node * HIDF + fc) = *(const float4*)(Dl + (long)m * HIDF + fc);
    }
}

// ---------------- GEMM2: [N,128] @ [128,128], TDM-staged, 4 K-steps ---------------
__global__ void gemm2_kernel(const __bf16* __restrict__ Abf, const __bf16* __restrict__ Wbf,
                             float* __restrict__ Hout, int n_nodes) {
    extern __shared__ char smem[];
    __bf16* Wl = (__bf16*)smem;             // 32768 B (128x128 swizzled)
    __bf16* Al = (__bf16*)(smem + 32768);   // 4096 B  (16x128 swizzled)
    float*  Dl = (float*)(smem + 36864);    // 8192 B
    const int tid = threadIdx.x;            // 128 threads = 4 waves
    const long node0 = (long)blockIdx.x * 16;
    if (tid < 32) {
        tdm_load_bf16(0u, Wbf, 16384u, 16384u);
        tdm_load_bf16(32768u, Abf + (long)blockIdx.x * 2048, 2048u, 2048u);
        __builtin_amdgcn_s_wait_tensorcnt((short)0);
    }
    __syncthreads();
    const int wave = tid >> 5, lane = tid & 31;
    const int half = lane >> 4, mn = lane & 15;
#pragma unroll
    for (int ct = 0; ct < 2; ++ct) {
        const int c0 = (wave * 2 + ct) * 16;
        v8f c = {};
#pragma unroll
        for (int kt = 0; kt < 4; ++kt) {
            v16bf a = *(const v16bf*)(Al + ((kt * 16 + mn) * 2 + half) * 16);
            v16bf b = *(const v16bf*)(Wl + ((kt * HIDF + c0 + mn) * 2 + half) * 16);
            c = __builtin_amdgcn_wmma_f32_16x16x32_bf16(false, a, false, b, (short)0, c, false, false);
        }
#pragma unroll
        for (int r = 0; r < 8; ++r) Dl[(r + 8 * half) * HIDF + c0 + mn] = c[r];
    }
    __syncthreads();
#pragma unroll
    for (int it = 0; it < 4; ++it) {
        int fi = it * 128 + tid;
        int m = fi >> 5, fc = (fi & 31) * 4;
        long node = node0 + m;
        if (node < n_nodes)
            *(float4*)(Hout + node * HIDF + fc) = *(const float4*)(Dl + (long)m * HIDF + fc);
    }
}

// ---- edge aggregation: LDS-staged edge records, prefetch, f32 atomics ------------
__global__ void aggregate_kernel(const float* __restrict__ Hin, const float* __restrict__ dinv,
                                 const int* __restrict__ row, const int* __restrict__ col,
                                 const float* __restrict__ w, float* __restrict__ Agg,
                                 long n_edges) {
    __shared__ int   sR[EPB];
    __shared__ int   sC[EPB];
    __shared__ float sN[EPB];
    const int tid = threadIdx.x;               // 256 threads
    const long e0 = (long)blockIdx.x * EPB;
    if (tid < EPB) {
        long e = e0 + tid;
        int r = 0, c = 0; float nm = 0.0f;
        if (e < n_edges) {
            r = row[e]; c = col[e];
            nm = dinv[r] * w[e] * dinv[c];
        }
        sR[tid] = r; sC[tid] = c; sN[tid] = nm;
    }
    __syncthreads();
#pragma unroll
    for (int it = 0; it < EPB * 32 / 256; ++it) {   // 8 iterations
        int slot = it * 256 + tid;
        int el = slot >> 5;
        int f  = (slot & 31) * 4;
        if (it + 1 < EPB * 32 / 256) {
            int eln = (slot + 256) >> 5;            // overlap next gather's latency
            __builtin_prefetch(Hin + (long)sR[eln] * HIDF + f, 0, 1);
        }
        float nm = sN[el];
        if (nm != 0.0f) {
            const float4 h = *(const float4*)(Hin + (long)sR[el] * HIDF + f);
            float* o = Agg + (long)sC[el] * HIDF + f;
            atomicAdd(o + 0, h.x * nm);
            atomicAdd(o + 1, h.y * nm);
            atomicAdd(o + 2, h.z * nm);
            atomicAdd(o + 3, h.w * nm);
        }
    }
}

// ---- layer1 fused: bf16(relu(Agg + Hx*dinv^2 + b1)) written pre-swizzled ---------
__global__ void sbr1_kernel(const float* __restrict__ Agg, const float* __restrict__ Hx,
                            const float* __restrict__ dinv, const float* __restrict__ bias,
                            __bf16* __restrict__ OutBf, int n_nodes) {
    long gid = (long)blockIdx.x * blockDim.x + threadIdx.x;
    long i = gid >> 5;
    if (i >= n_nodes) return;
    int f = (int)(gid & 31) * 4;
    float d = dinv[i]; float d2 = d * d;
    float4 a  = *(const float4*)(Agg + i * HIDF + f);
    float4 hx = *(const float4*)(Hx  + i * HIDF + f);
    float o0 = fmaxf(a.x + hx.x * d2 + bias[f + 0], 0.f);
    float o1 = fmaxf(a.y + hx.y * d2 + bias[f + 1], 0.f);
    float o2 = fmaxf(a.z + hx.z * d2 + bias[f + 2], 0.f);
    float o3 = fmaxf(a.w + hx.w * d2 + bias[f + 3], 0.f);
    // swizzled A-fragment position (matches gemm2 Al layout)
    long tile = i >> 4; int m = (int)(i & 15);
    int kt = f >> 5, kk = f & 31;
    int h = (kk >> 3) & 1, j = (kk & 7) + 8 * (kk >> 4);
    unsigned* p = (unsigned*)(OutBf + tile * 2048 + ((kt * 16 + m) * 2 + h) * 16 + j);
    p[0] = (unsigned)f2bfbits(o0) | ((unsigned)f2bfbits(o1) << 16);
    p[1] = (unsigned)f2bfbits(o2) | ((unsigned)f2bfbits(o3) << 16);
}

// ---- layer2 fused: f32 (Agg + Hx*dinv^2 + b2), written in place ------------------
__global__ void sbr2_kernel(const float* __restrict__ Agg, const float* __restrict__ Hx,
                            const float* __restrict__ dinv, const float* __restrict__ bias,
                            float* __restrict__ Out, int n_nodes) {
    long gid = (long)blockIdx.x * blockDim.x + threadIdx.x;
    long i = gid >> 5;
    if (i >= n_nodes) return;
    int f = (int)(gid & 31) * 4;
    float d = dinv[i]; float d2 = d * d;
    float4 a  = *(const float4*)(Agg + i * HIDF + f);
    float4 hx = *(const float4*)(Hx  + i * HIDF + f);
    float4 o;
    o.x = a.x + hx.x * d2 + bias[f + 0];
    o.y = a.y + hx.y * d2 + bias[f + 1];
    o.z = a.z + hx.z * d2 + bias[f + 2];
    o.w = a.w + hx.w * d2 + bias[f + 3];
    *(float4*)(Out + i * HIDF + f) = o;
}

// ---------------- global mean pool + head -----------------------------------------
__global__ void pool_kernel(const float* __restrict__ H, const int* __restrict__ batch,
                            float* __restrict__ sums, float* __restrict__ cnts, int n_nodes) {
    long gid = (long)blockIdx.x * blockDim.x + threadIdx.x;
    long i = gid >> 5;
    if (i >= n_nodes) return;
    int f = (int)(gid & 31) * 4;
    int g = batch[i];
    float4 h = *(const float4*)(H + i * HIDF + f);
    float* s = sums + (long)g * HIDF + f;
    atomicAdd(s + 0, h.x); atomicAdd(s + 1, h.y);
    atomicAdd(s + 2, h.z); atomicAdd(s + 3, h.w);
    if (f == 0) atomicAdd(&cnts[g], 1.0f);
}

__global__ void final_kernel(const float* __restrict__ sums, const float* __restrict__ cnts,
                             const float* __restrict__ Wlin, const float* __restrict__ blin,
                             float* __restrict__ out) {
    int g = blockIdx.x * blockDim.x + threadIdx.x;
    if (g >= NGRAPH) return;
    float acc = 0.0f;
#pragma unroll 4
    for (int f = 0; f < HIDF; ++f) acc += sums[(long)g * HIDF + f] * Wlin[f];
    out[g] = acc / fmaxf(cnts[g], 1.0f) + blin[0];
}

// ----------------------------------------------------------------------------------
extern "C" void kernel_launch(void* const* d_in, const int* in_sizes, int n_in,
                              void* d_out, int out_size, void* d_ws, size_t ws_size,
                              hipStream_t stream) {
    const float* x    = (const float*)d_in[0];
    const int*   ei   = (const int*)d_in[1];
    const float* ew   = (const float*)d_in[2];
    const int*   bat  = (const int*)d_in[3];
    const float* W1   = (const float*)d_in[4];
    const float* b1   = (const float*)d_in[5];
    const float* W2   = (const float*)d_in[6];
    const float* b2   = (const float*)d_in[7];
    const float* Wlin = (const float*)d_in[8];
    const float* blin = (const float*)d_in[9];
    float* out = (float*)d_out;

    const int  n_nodes = in_sizes[0] / N_FIN;
    const long n_edges = (long)in_sizes[1] / 2;
    const int  ntiles  = (n_nodes + 15) / 16;
    const int* row = ei;
    const int* col = ei + n_edges;

    // workspace layout (16B aligned pieces)
    char* ws = (char*)d_ws;
    size_t off = 0;
    float*  deg  = (float*)(ws + off);  off += (size_t)((n_nodes + 3) & ~3) * 4;
    float*  dinv = (float*)(ws + off);  off += (size_t)((n_nodes + 3) & ~3) * 4;
    float*  bufA = (float*)(ws + off);  off += (size_t)ntiles * 16 * HIDF * 4; // GEMM outs
    float*  bufC = (float*)(ws + off);  off += (size_t)ntiles * 16 * HIDF * 4; // agg target
    __bf16* xp   = (__bf16*)(ws + off); off += (size_t)ntiles * 512 * 2;       // swizzled x
    __bf16* h1bf = (__bf16*)(ws + off); off += (size_t)ntiles * 2048 * 2;      // swizzled h1
    __bf16* w1p  = (__bf16*)(ws + off); off += 4096 * 2;                       // swizzled W1
    __bf16* w2bf = (__bf16*)(ws + off); off += 16384 * 2;                      // swizzled W2
    float*  sums = (float*)(ws + off);  off += (size_t)NGRAPH * HIDF * 4;
    float*  cnts = (float*)(ws + off);  off += NGRAPH * 4;

    const int T = 256;
    const int nodef_blocks = (int)(((long)n_nodes * 32 + T - 1) / T);
    const int agg_blocks   = (int)((n_edges + EPB - 1) / EPB);

    // degrees & symmetric norm
    (void)hipMemsetAsync(deg, 0, (size_t)n_nodes * 4, stream);
    degree_kernel<<<(int)((n_edges + T - 1) / T), T, 0, stream>>>(col, ew, deg, n_edges);
    dinv_kernel<<<(n_nodes + T - 1) / T, T, 0, stream>>>(deg, dinv, n_nodes);

    // operand preparation (bf16, WMMA-swizzled)
    (void)hipMemsetAsync(xp, 0, (size_t)ntiles * 512 * 2, stream);
    convw_kernel<<<(32 * HIDF + T - 1) / T, T, 0, stream>>>(W1, w1p, N_FIN, 32);
    convw_kernel<<<(HIDF * HIDF + T - 1) / T, T, 0, stream>>>(W2, w2bf, HIDF, HIDF);
    convx_kernel<<<nodef_blocks, T, 0, stream>>>(x, xp, n_nodes);

    // layer 1
    gemm1_kernel<<<ntiles, 128, 17408, stream>>>(xp, w1p, bufA, n_nodes);
    (void)hipMemsetAsync(bufC, 0, (size_t)n_nodes * HIDF * 4, stream);
    aggregate_kernel<<<agg_blocks, T, 0, stream>>>(bufA, dinv, row, col, ew, bufC, n_edges);
    (void)hipMemsetAsync(h1bf, 0, (size_t)ntiles * 2048 * 2, stream);
    sbr1_kernel<<<nodef_blocks, T, 0, stream>>>(bufC, bufA, dinv, b1, h1bf, n_nodes);

    // layer 2
    gemm2_kernel<<<ntiles, 128, 45056, stream>>>(h1bf, w2bf, bufA, n_nodes);
    (void)hipMemsetAsync(bufC, 0, (size_t)n_nodes * HIDF * 4, stream);
    aggregate_kernel<<<agg_blocks, T, 0, stream>>>(bufA, dinv, row, col, ew, bufC, n_edges);
    sbr2_kernel<<<nodef_blocks, T, 0, stream>>>(bufC, bufA, dinv, b2, bufA, n_nodes);

    // pool + linear head
    (void)hipMemsetAsync(sums, 0, ((size_t)NGRAPH * HIDF + NGRAPH) * 4, stream);
    pool_kernel<<<nodef_blocks, T, 0, stream>>>(bufA, bat, sums, cnts, n_nodes);
    final_kernel<<<1, NGRAPH, 0, stream>>>(sums, cnts, Wlin, blin, out);
}
